// HyperGKAN_33191507264125
// MI455X (gfx1250) — matrix operand order repaired and checked by reading, not compiled, not run on hardware
//
#include <hip/hip_runtime.h>
#include <hip/hip_bf16.h>

// ---------------------------------------------------------------------------
// HyperGKAN on MI455X (gfx1250, wave32, WMMA f32<=f16).
// - All big matmuls via v_wmma_f32_16x16x32_f16 (f32 accumulate).
// - Hypergraph convs dense-ified (H_nei 36MB stays resident in the 192MB L2).
// - Spline features (295MB if materialized) generated in-register per fragment.
// - Fragment loads: clamped indices (no exec-mask ladders), float4/b128 wide.
// - Spline GEMM stages Ws (128KB) in LDS, via TDM tensor_load_to_lds when the
//   builtin is available (s_wait_tensorcnt), else cooperative b128 copy.
// ---------------------------------------------------------------------------

typedef __attribute__((ext_vector_type(16))) _Float16 v16h;
typedef __attribute__((ext_vector_type(8)))  float    v8f;
typedef __attribute__((ext_vector_type(4)))  unsigned v4u;
typedef __attribute__((ext_vector_type(8)))  int      v8i;
typedef __attribute__((ext_vector_type(4)))  int      v4i;

#define HB   4
#define HT   12
#define HN   3000
#define HE1  3000
#define HE2  384
#define NTOK (HB*HT*HN)   // 144000 rows of the (B,T,N,C) activation
#define NSEQ (HB*HN)      // 12000 GRU sequences

__device__ __forceinline__ float sigf(float x) { return 1.0f / (1.0f + __expf(-x)); }

// ---------------------------------------------------------------------------
// Generic WMMA GEMM: C[M,N] = op(A)[M,K] @ op(B)[K,N] (+bias[N]) (*rowscale[M])
//   AT:    A stored K-major (A[k*lda+m])  -- H^T in hconv einsum 1
//   BT:    B stored (N,K) row-major (weights)
//   ASILU: silu applied to A on the fly (KAN base path)
//   ACC:   accumulate onto existing C
// One wave -> 16 rows x 64 cols (4 tiles, 32 f32 acc VGPRs). 256 thr = 8 waves.
// OOB rows/cols are index-clamped (their results are simply never stored);
// only the K-tail needs value masking (single extra iteration).
// ---------------------------------------------------------------------------
template<bool AT, bool BT, bool ASILU, bool ACC>
__global__ __launch_bounds__(256)
void gemm_wmma(const float* __restrict__ A, const float* __restrict__ Bm,
               const float* __restrict__ bias, const float* __restrict__ rowscale,
               float* __restrict__ C,
               int M, int N, int K, int lda, int ldb, int ldc,
               long strideA, long strideB, long strideC)
{
    const int lane = threadIdx.x & 31;
    const int wave = threadIdx.x >> 5;
    const int row0 = (blockIdx.x * 8 + wave) * 16;
    const int col0 = blockIdx.y * 64;
    A  += (long)blockIdx.z * strideA;
    Bm += (long)blockIdx.z * strideB;
    C  += (long)blockIdx.z * strideC;
    if (row0 >= M) return;

    const int half = lane >> 4;
    const int mrow = min(row0 + (lane & 15), M - 1);   // clamped: OOB rows never stored
    int ncl[4];
#pragma unroll
    for (int t = 0; t < 4; ++t) ncl[t] = min(col0 + 16 * t + (lane & 15), N - 1);

    const bool fullM = (row0 + 16 <= M);
    const bool fullN = (col0 + 64 <= N);

    v8f acc[4];
    if (ACC) {
#pragma unroll
        for (int t = 0; t < 4; ++t)
#pragma unroll
            for (int r = 0; r < 8; ++r) {
                int rr = row0 + r + 8 * half;
                int cc = col0 + 16 * t + (lane & 15);
                float v = C[(long)min(rr, M - 1) * ldc + min(cc, N - 1)];
                acc[t][r] = (rr < M && cc < N) ? v : 0.0f;
            }
    } else {
#pragma unroll
        for (int t = 0; t < 4; ++t)
#pragma unroll
            for (int r = 0; r < 8; ++r) acc[t][r] = 0.0f;
    }

    const int Kmain = K & ~31;
    int k0 = 0;
    for (; k0 < Kmain; k0 += 32) {
        if (k0 + 32 < K) {   // gfx1250 global_prefetch_b8
            const float* pf = AT ? (A + (long)(k0 + 32) * lda + mrow)
                                 : (A + (long)mrow * lda + (k0 + 32));
            __builtin_prefetch(pf, 0, 3);
        }
        // ---- A fragment (16x32 f16): lane row = mrow; ISA k layout:
        //      k = (j<4 ? 2j : 16+2(j-4)) + 8*half  -> two 8-float runs
        v16h af;
        if (AT) {
#pragma unroll
            for (int j = 0; j < 8; ++j) {
                int kb = ((j < 4) ? 2 * j : 16 + 2 * (j - 4)) + 8 * half + k0;
                float a0 = A[(long)kb * lda + mrow];
                float a1 = A[(long)(kb + 1) * lda + mrow];
                if (ASILU) { a0 *= sigf(a0); a1 *= sigf(a1); }
                af[2 * j] = (_Float16)a0; af[2 * j + 1] = (_Float16)a1;
            }
        } else {
            const float* ap = A + (long)mrow * lda + k0 + 8 * half;
            float4 q0 = *(const float4*)(ap);
            float4 q1 = *(const float4*)(ap + 4);
            float4 q2 = *(const float4*)(ap + 16);
            float4 q3 = *(const float4*)(ap + 20);
            float av[16] = { q0.x, q0.y, q0.z, q0.w, q1.x, q1.y, q1.z, q1.w,
                             q2.x, q2.y, q2.z, q2.w, q3.x, q3.y, q3.z, q3.w };
#pragma unroll
            for (int i = 0; i < 16; ++i) {
                float a = av[i];
                if (ASILU) a *= sigf(a);
                af[i] = (_Float16)a;
            }
        }
        // ---- 4 output tiles share the A fragment
#pragma unroll
        for (int t = 0; t < 4; ++t) {
            v16h bf;
            if (BT) {
                // lane's B fragment = two 8-float runs (k = 8*half + {0..7, 16..23})
                const float* bp = Bm + (long)ncl[t] * ldb + k0 + 8 * half;
                float4 q0 = *(const float4*)(bp);
                float4 q1 = *(const float4*)(bp + 4);
                float4 q2 = *(const float4*)(bp + 16);
                float4 q3 = *(const float4*)(bp + 20);
                float bv[16] = { q0.x, q0.y, q0.z, q0.w, q1.x, q1.y, q1.z, q1.w,
                                 q2.x, q2.y, q2.z, q2.w, q3.x, q3.y, q3.z, q3.w };
#pragma unroll
                for (int i = 0; i < 16; ++i) bf[i] = (_Float16)bv[i];
            } else {
#pragma unroll
                for (int j = 0; j < 8; ++j) {
                    int kb = ((j < 4) ? 2 * j : 16 + 2 * (j - 4)) + 8 * half + k0;
                    bf[2 * j]     = (_Float16)Bm[(long)kb * ldb + ncl[t]];
                    bf[2 * j + 1] = (_Float16)Bm[(long)(kb + 1) * ldb + ncl[t]];
                }
            }
            acc[t] = __builtin_amdgcn_wmma_f32_16x16x32_f16(
                false, af, false, bf, (short)0, acc[t], false, false);
        }
    }

    // ---- K tail (only K=3000 hits this): clamped address + value mask
    if (k0 < K) {
        v16h af;
#pragma unroll
        for (int j = 0; j < 8; ++j) {
            int kb = ((j < 4) ? 2 * j : 16 + 2 * (j - 4)) + 8 * half + k0;
            int kc0 = min(kb, K - 1), kc1 = min(kb + 1, K - 1);
            float a0 = AT ? A[(long)kc0 * lda + mrow] : A[(long)mrow * lda + kc0];
            float a1 = AT ? A[(long)kc1 * lda + mrow] : A[(long)mrow * lda + kc1];
            a0 = (kb     < K) ? a0 : 0.0f;
            a1 = (kb + 1 < K) ? a1 : 0.0f;
            if (ASILU) { a0 *= sigf(a0); a1 *= sigf(a1); }
            af[2 * j] = (_Float16)a0; af[2 * j + 1] = (_Float16)a1;
        }
#pragma unroll
        for (int t = 0; t < 4; ++t) {
            v16h bf;
#pragma unroll
            for (int j = 0; j < 8; ++j) {
                int kb = ((j < 4) ? 2 * j : 16 + 2 * (j - 4)) + 8 * half + k0;
                int kc0 = min(kb, K - 1), kc1 = min(kb + 1, K - 1);
                float b0 = BT ? Bm[(long)ncl[t] * ldb + kc0] : Bm[(long)kc0 * ldb + ncl[t]];
                float b1 = BT ? Bm[(long)ncl[t] * ldb + kc1] : Bm[(long)kc1 * ldb + ncl[t]];
                b0 = (kb     < K) ? b0 : 0.0f;
                b1 = (kb + 1 < K) ? b1 : 0.0f;
                bf[2 * j] = (_Float16)b0; bf[2 * j + 1] = (_Float16)b1;
            }
            acc[t] = __builtin_amdgcn_wmma_f32_16x16x32_f16(
                false, af, false, bf, (short)0, acc[t], false, false);
        }
    }

    // ---- epilogue
    if (fullM && fullN) {
#pragma unroll
        for (int t = 0; t < 4; ++t)
#pragma unroll
            for (int r = 0; r < 8; ++r) {
                int rr = row0 + r + 8 * half;
                int cc = col0 + 16 * t + (lane & 15);
                float v = acc[t][r];
                if (bias)     v += bias[cc];
                if (rowscale) v *= rowscale[rr];
                C[(long)rr * ldc + cc] = v;
            }
    } else {
#pragma unroll
        for (int t = 0; t < 4; ++t)
#pragma unroll
            for (int r = 0; r < 8; ++r) {
                int rr = row0 + r + 8 * half;
                int cc = col0 + 16 * t + (lane & 15);
                if (rr < M && cc < N) {
                    float v = acc[t][r];
                    if (bias)     v += bias[cc];
                    if (rowscale) v *= rowscale[rr];
                    C[(long)rr * ldc + cc] = v;
                }
            }
    }
}

// ---------------------------------------------------------------------------
// Cubic B-spline bases on the uniform KAN grid: 8 bases from 12 knots.
// ---------------------------------------------------------------------------
__device__ __forceinline__ void bspline8(float x, float* w8)
{
    const float h = 2.0f / 5.0f;          // GRID=5 on [-1,1]
    float g[12];
#pragma unroll
    for (int t = 0; t < 12; ++t) g[t] = -1.0f + (float)(t - 3) * h;
    float w[11];
#pragma unroll
    for (int i = 0; i < 11; ++i) w[i] = (x >= g[i] && x < g[i + 1]) ? 1.0f : 0.0f;
#pragma unroll
    for (int j = 1; j <= 3; ++j) {
#pragma unroll
        for (int i = 0; i + j < 11; ++i) {
            w[i] = (x - g[i]) / (g[i + j] - g[i]) * w[i]
                 + (g[i + j + 1] - x) / (g[i + j + 1] - g[i + 1]) * w[i + 1];
        }
    }
#pragma unroll
    for (int i = 0; i < 8; ++i) w8[i] = w[i];
}

// ---------------------------------------------------------------------------
// KAN spline GEMM: C[M,64] += bases(Xin)[M,512] @ Ws(64,512)^T
// Ws (128KB f32) is staged once per workgroup into LDS (WGP has 320KB);
// each lane then needs only 2 channels' bases per 32-wide k-step of A.
// Grid exactly covers M (1125 blocks * 8 waves * 16 rows = 144000): no wave
// exits before the barrier.
// ---------------------------------------------------------------------------
__global__ __launch_bounds__(256)
void kan_spline_gemm(const float* __restrict__ Xin, const float* __restrict__ Ws,
                     float* __restrict__ C, int M)
{
    __shared__ float lws[64 * 512];       // 128 KB
    const int lane = threadIdx.x & 31;
    const int wave = threadIdx.x >> 5;

#if __has_builtin(__builtin_amdgcn_tensor_load_to_lds)
    // Tensor Data Mover: 2D tile 512x64 f32 global -> LDS (D# per ISA ch.8)
    if (wave == 0) {
        unsigned lds_off = (unsigned)(unsigned long long)(__SIZE_TYPE__)lws;
        unsigned long long ga = (unsigned long long)(__SIZE_TYPE__)Ws;
        v4u g0 = { 1u,                                   // count=1, user mode
                   lds_off,                              // lds_addr
                   (unsigned)(ga & 0xFFFFFFFFu),         // global_addr[31:0]
                   (unsigned)((ga >> 32) & 0x1FFFFFFu) | (2u << 30) }; // [56:32] | type=2
        v8i g1 = { (int)(2u << 16),                      // data_size=4B
                   (int)(512u << 16),                    // tensor_dim0=512 (lo16 @63:48)
                   (int)(64u << 16),                     // tensor_dim1=64  (lo16 @95:80)
                   (int)(512u << 16),                    // tile_dim0=512   (@127:112)
                   64,                                   // tile_dim1=64    (@143:128)
                   512,                                  // tensor_dim0_stride=512
                   0, 0 };
        v4i z4 = { 0, 0, 0, 0 };
#if defined(__clang_major__) && (__clang_major__ >= 23)
        v8i z8 = { 0, 0, 0, 0, 0, 0, 0, 0 };
        __builtin_amdgcn_tensor_load_to_lds(g0, g1, z4, z4, z8, 0);
#else
        __builtin_amdgcn_tensor_load_to_lds(g0, g1, z4, z4, 0);
#endif
        __builtin_amdgcn_s_wait_tensorcnt(0);
    }
    __syncthreads();
#else
    for (int i = threadIdx.x * 4; i < 64 * 512; i += 256 * 4)
        *(float4*)&lws[i] = *(const float4*)&Ws[i];
    __syncthreads();
#endif

    const int row0 = (blockIdx.x * 8 + wave) * 16;
    const int mrow = row0 + (lane & 15);
    const int half = lane >> 4;

    v8f acc[4];
#pragma unroll
    for (int t = 0; t < 4; ++t)
#pragma unroll
        for (int r = 0; r < 8; ++r) {
            int rr = row0 + r + 8 * half;
            int cc = 16 * t + (lane & 15);
            acc[t][r] = C[(long)rr * 64 + cc];     // accumulate (M tiles exact)
        }

    for (int k0 = 0; k0 < 512; k0 += 32) {
        // k = c*8 + basis; this fragment touches channels c0 (j<4) and c0+2 (j>=4)
        int c0 = (k0 >> 3) + half;
        int c1 = c0 + 2;
        float x0 = Xin[(long)mrow * 64 + c0];
        float x1 = Xin[(long)mrow * 64 + c1];
        float bs0[8], bs1[8];
        bspline8(x0, bs0);
        bspline8(x1, bs1);

        v16h af;
#pragma unroll
        for (int j = 0; j < 8; ++j) {
            int jj = 2 * (j & 3);
            float e0 = (j < 4) ? bs0[jj]     : bs1[jj];
            float e1 = (j < 4) ? bs0[jj + 1] : bs1[jj + 1];
            af[2 * j]     = (_Float16)e0;
            af[2 * j + 1] = (_Float16)e1;
        }
#pragma unroll
        for (int t = 0; t < 4; ++t) {
            int ncol = 16 * t + (lane & 15);
            // lane's B fragment = two 8-f32 runs in LDS (k = 8*half + {0..7, 16..23}) -> ds_load_b128 x4
            const float* bp = &lws[ncol * 512 + k0 + 8 * half];
            float4 q0 = *(const float4*)(bp);
            float4 q1 = *(const float4*)(bp + 4);
            float4 q2 = *(const float4*)(bp + 16);
            float4 q3 = *(const float4*)(bp + 20);
            float bv[16] = { q0.x, q0.y, q0.z, q0.w, q1.x, q1.y, q1.z, q1.w,
                             q2.x, q2.y, q2.z, q2.w, q3.x, q3.y, q3.z, q3.w };
            v16h bf;
#pragma unroll
            for (int i = 0; i < 16; ++i) bf[i] = (_Float16)bv[i];
            acc[t] = __builtin_amdgcn_wmma_f32_16x16x32_f16(
                false, af, false, bf, (short)0, acc[t], false, false);
        }
    }

#pragma unroll
    for (int t = 0; t < 4; ++t)
#pragma unroll
        for (int r = 0; r < 8; ++r) {
            int rr = row0 + r + 8 * half;
            int cc = 16 * t + (lane & 15);
            C[(long)rr * 64 + cc] = acc[t][r];
        }
}

// ---------------------------------------------------------------------------
// Elementwise / reduction kernels
// ---------------------------------------------------------------------------
__global__ void zero_f32(float* p, long n)
{
    long i = (long)blockIdx.x * 256 + threadIdx.x;
    if (i < n) p[i] = 0.0f;
}

__global__ void in_proj(const float* __restrict__ x, const float* __restrict__ W,
                        const float* __restrict__ b, float* __restrict__ X)
{
    long i = (long)blockIdx.x * 256 + threadIdx.x;      // over NTOK*64
    if (i >= (long)NTOK * 64) return;
    int  c = (int)(i & 63);
    long r = i >> 6;
    X[i] = x[r * 2] * W[c * 2] + x[r * 2 + 1] * W[c * 2 + 1] + b[c];
}

// s1[e] = W[e] / clip(sum_n H[n][e])
__global__ void edge_scale(const float* __restrict__ H, const float* __restrict__ W,
                           float* __restrict__ s1, int N, int E)
{
    int e = blockIdx.x * 256 + threadIdx.x;
    if (e >= E) return;
    float de = 0.0f;
    for (int n = 0; n < N; ++n) de += H[(long)n * E + e];
    s1[e] = W[e] / fmaxf(de, 1e-6f);
}

// s2[n] = 1 / clip(sum_e H[n][e]*W[e])
__global__ void node_scale(const float* __restrict__ H, const float* __restrict__ W,
                           float* __restrict__ s2, int N, int E)
{
    int n = blockIdx.x * 256 + threadIdx.x;
    if (n >= N) return;
    float dv = 0.0f;
    for (int e = 0; e < E; ++e) dv += H[(long)n * E + e] * W[e];
    s2[n] = 1.0f / fmaxf(dv, 1e-6f);
}

// Out = LN(F + R) * gamma + beta ; one wave32 per 64-channel row, in-place safe.
__global__ __launch_bounds__(256)
void ln_residual(const float* __restrict__ F, const float* __restrict__ R,
                 const float* __restrict__ gamma, const float* __restrict__ beta,
                 float* __restrict__ Out, int rows)
{
    int lane = threadIdx.x & 31;
    int wave = threadIdx.x >> 5;
    int r = blockIdx.x * 8 + wave;
    if (r >= rows) return;
    long base = (long)r * 64;
    float v0 = F[base + lane]      + R[base + lane];
    float v1 = F[base + 32 + lane] + R[base + 32 + lane];
    float s = v0 + v1;
#pragma unroll
    for (int o = 16; o > 0; o >>= 1) s += __shfl_xor(s, o, 32);
    float mu = s * (1.0f / 64.0f);
    float d0 = v0 - mu, d1 = v1 - mu;
    float q = d0 * d0 + d1 * d1;
#pragma unroll
    for (int o = 16; o > 0; o >>= 1) q += __shfl_xor(q, o, 32);
    float inv = rsqrtf(q * (1.0f / 64.0f) + 1e-5f);
    Out[base + lane]      = d0 * inv * gamma[lane]      + beta[lane];
    Out[base + 32 + lane] = d1 * inv * gamma[32 + lane] + beta[32 + lane];
}

// (B,T,N,64) -> (B,N,T,64)
__global__ void transpose_btnc(const float* __restrict__ X, float* __restrict__ XR)
{
    long i = (long)blockIdx.x * 256 + threadIdx.x;
    if (i >= (long)NTOK * 64) return;
    int  c = (int)(i & 63);
    long r = i >> 6;
    int  n = (int)(r % HN);
    long bt = r / HN;
    int  t = (int)(bt % HT);
    int  b = (int)(bt / HT);
    XR[(((long)(b * HN + n)) * HT + t) * 64 + c] = X[i];
}

// X[:,t,:,:] = X[:,T-1,:,:] for t != T-1 (readers never touch written slots)
__global__ void bcast_last(float* __restrict__ X)
{
    long i = (long)blockIdx.x * 256 + threadIdx.x;
    if (i >= (long)NTOK * 64) return;
    int  c = (int)(i & 63);
    long r = i >> 6;
    int  n = (int)(r % HN);
    long bt = r / HN;
    int  t = (int)(bt % HT);
    int  b = (int)(bt / HT);
    if (t == HT - 1) return;
    X[i] = X[(((long)b * HT + (HT - 1)) * HN + n) * 64 + c];
}

// GRU gate update: h = (1-z)*n + z*h, also writes the sequence output slot.
__global__ void gru_gate(const float* __restrict__ gi, const float* __restrict__ gh,
                         float* __restrict__ h, float* __restrict__ seq,
                         int S, int t, int T)
{
    int i = blockIdx.x * 256 + threadIdx.x;             // over S*64
    if (i >= S * 64) return;
    int c = i & 63;
    int s = i >> 6;
    const float* pi = gi + (long)s * 192;
    const float* ph = gh + (long)s * 192;
    float r = sigf(pi[c] + ph[c]);
    float z = sigf(pi[64 + c] + ph[64 + c]);
    float n = tanhf(pi[128 + c] + r * ph[128 + c]);
    float hn = (1.0f - z) * n + z * h[i];
    h[i] = hn;
    seq[((long)s * T + t) * 64 + c] = hn;
}

// out[b,t,n] = dot64(seq[(b,n,t)], out_W) + out_b
__global__ void out_proj(const float* __restrict__ SEQ, const float* __restrict__ W,
                         const float* __restrict__ b, float* __restrict__ out)
{
    long i = (long)blockIdx.x * 256 + threadIdx.x;      // over NTOK
    if (i >= NTOK) return;
    int  n = (int)(i % HN);
    long bt = i / HN;
    int  t  = (int)(bt % HT);
    int  bb = (int)(bt / HT);
    const float* row = SEQ + (((long)(bb * HN + n)) * HT + t) * 64;
    float acc = b[0];
#pragma unroll
    for (int c = 0; c < 64; ++c) acc += row[c] * W[c];
    out[i] = acc;
}

// ---------------------------------------------------------------------------
// Host-side orchestration
// ---------------------------------------------------------------------------
struct Lay  { const float *Wf, *bf, *beta, *gamma, *WbN, *WsN, *WbS, *WsS; };
struct GruL { const float *Wh, *Wi, *bh, *bi; };

static Lay layAt(void* const* d, int b)
{
    Lay L;
    L.Wf    = (const float*)d[b];     L.bf  = (const float*)d[b + 1];
    L.beta  = (const float*)d[b + 2]; L.gamma = (const float*)d[b + 3];
    L.WbN   = (const float*)d[b + 4]; L.WsN = (const float*)d[b + 5];   // b+6 = grid (uniform, unused)
    L.WbS   = (const float*)d[b + 7]; L.WsS = (const float*)d[b + 8];   // b+9 = grid
    return L;
}
static GruL gruAt(void* const* d, int b)
{
    GruL g;
    g.Wh = (const float*)d[b];     g.Wi = (const float*)d[b + 1];
    g.bh = (const float*)d[b + 2]; g.bi = (const float*)d[b + 3];
    return g;
}

static void run_layer(const Lay& L,
                      const float* H_nei, const float* s1n, const float* s2n,
                      const float* H_sem, const float* s1s, const float* s2s,
                      float* X, float* T1, float* HX, float* ON, float* OS,
                      hipStream_t stream)
{
    dim3 blk(256);
    const long szN  = (long)HN  * 64;   // 192000 floats per (N,64) slice
    const long szE2 = (long)HE2 * 64;   // 24576
    // --- hconv (nei): tmp[m] = s1 .* (H^T @ x_m) ; hx[m] = s2 .* (H @ tmp[m])
    gemm_wmma<true, false, false, false><<<dim3(24, 1, 48), blk, 0, stream>>>(
        H_nei, X, nullptr, s1n, T1, HE1, 64, HN, HE1, 64, 64, 0L, szN, szN);
    gemm_wmma<false, false, false, false><<<dim3(24, 1, 48), blk, 0, stream>>>(
        H_nei, T1, nullptr, s2n, HX, HN, 64, HE1, HE1, 64, 64, 0L, szN, szN);
    // --- KAN (nei): ON = silu(HX)@Wb^T  then  += spline(HX)@Ws^T
    gemm_wmma<false, true, true, false><<<dim3(1125, 1, 1), blk, 0, stream>>>(
        HX, L.WbN, nullptr, nullptr, ON, NTOK, 64, 64, 64, 64, 64, 0L, 0L, 0L);
    kan_spline_gemm<<<dim3(1125), blk, 0, stream>>>(HX, L.WsN, ON, NTOK);
    // --- hconv (sem)
    gemm_wmma<true, false, false, false><<<dim3(3, 1, 48), blk, 0, stream>>>(
        H_sem, X, nullptr, s1s, T1, HE2, 64, HN, HE2, 64, 64, 0L, szN, szE2);
    gemm_wmma<false, false, false, false><<<dim3(24, 1, 48), blk, 0, stream>>>(
        H_sem, T1, nullptr, s2s, HX, HN, 64, HE2, HE2, 64, 64, 0L, szE2, szN);
    // --- KAN (sem)
    gemm_wmma<false, true, true, false><<<dim3(1125, 1, 1), blk, 0, stream>>>(
        HX, L.WbS, nullptr, nullptr, OS, NTOK, 64, 64, 64, 64, 64, 0L, 0L, 0L);
    kan_spline_gemm<<<dim3(1125), blk, 0, stream>>>(HX, L.WsS, OS, NTOK);
    // --- fuse: T1 = [ON|OS] @ W_fuse^T + b  (split K=128 into two K=64 GEMMs)
    gemm_wmma<false, true, false, false><<<dim3(1125, 1, 1), blk, 0, stream>>>(
        ON, L.Wf, L.bf, nullptr, T1, NTOK, 64, 64, 64, 128, 64, 0L, 0L, 0L);
    gemm_wmma<false, true, false, true><<<dim3(1125, 1, 1), blk, 0, stream>>>(
        OS, L.Wf + 64, nullptr, nullptr, T1, NTOK, 64, 64, 64, 128, 64, 0L, 0L, 0L);
    // --- residual + LayerNorm back into X
    ln_residual<<<dim3(18000), blk, 0, stream>>>(T1, X, L.gamma, L.beta, X, NTOK);
}

static void run_gru2(const GruL& g0, const GruL& g1,
                     const float* XR, float* seq0, float* seq1,
                     float* h0, float* h1, float* gi, float* gh,
                     bool zero_h, hipStream_t stream)
{
    dim3 blk(256);
    if (zero_h) {
        zero_f32<<<dim3(3000), blk, 0, stream>>>(h0, (long)NSEQ * 64);
        zero_f32<<<dim3(3000), blk, 0, stream>>>(h1, (long)NSEQ * 64);
    }
    for (int t = 0; t < HT; ++t) {     // layer 0 over full sequence
        gemm_wmma<false, true, false, false><<<dim3(94, 3, 1), blk, 0, stream>>>(
            XR + t * 64, g0.Wi, g0.bi, nullptr, gi, NSEQ, 192, 64, HT * 64, 64, 192, 0L, 0L, 0L);
        gemm_wmma<false, true, false, false><<<dim3(94, 3, 1), blk, 0, stream>>>(
            h0, g0.Wh, g0.bh, nullptr, gh, NSEQ, 192, 64, 64, 64, 192, 0L, 0L, 0L);
        gru_gate<<<dim3(3000), blk, 0, stream>>>(gi, gh, h0, seq0, NSEQ, t, HT);
    }
    for (int t = 0; t < HT; ++t) {     // layer 1
        gemm_wmma<false, true, false, false><<<dim3(94, 3, 1), blk, 0, stream>>>(
            seq0 + t * 64, g1.Wi, g1.bi, nullptr, gi, NSEQ, 192, 64, HT * 64, 64, 192, 0L, 0L, 0L);
        gemm_wmma<false, true, false, false><<<dim3(94, 3, 1), blk, 0, stream>>>(
            h1, g1.Wh, g1.bh, nullptr, gh, NSEQ, 192, 64, 64, 64, 192, 0L, 0L, 0L);
        gru_gate<<<dim3(3000), blk, 0, stream>>>(gi, gh, h1, seq1, NSEQ, t, HT);
    }
}

extern "C" void kernel_launch(void* const* d_in, const int* in_sizes, int n_in,
                              void* d_out, int out_size, void* d_ws, size_t ws_size,
                              hipStream_t stream)
{
    (void)in_sizes; (void)n_in; (void)out_size; (void)ws_size;
    // pytree-flattened input order (dicts sorted alphabetically):
    // 0:x 1:H_nei 2:H_sem 3:W_nei 4:W_sem
    // 5..24 dec layers, 25..32 dec_gru, 33..52 enc layers, 53..60 enc_gru,
    // 61:in_W 62:in_b 63:out_W 64:out_b
    const float* x     = (const float*)d_in[0];
    const float* H_nei = (const float*)d_in[1];
    const float* H_sem = (const float*)d_in[2];
    const float* W_nei = (const float*)d_in[3];
    const float* W_sem = (const float*)d_in[4];
    Lay  dec0 = layAt(d_in, 5),  dec1 = layAt(d_in, 15);
    GruL dg0  = gruAt(d_in, 25), dg1  = gruAt(d_in, 29);
    Lay  enc0 = layAt(d_in, 33), enc1 = layAt(d_in, 43);
    GruL eg0  = gruAt(d_in, 53), eg1  = gruAt(d_in, 57);
    const float* in_W  = (const float*)d_in[61];
    const float* in_b  = (const float*)d_in[62];
    const float* out_W = (const float*)d_in[63];
    const float* out_b = (const float*)d_in[64];

    // workspace layout (floats); total ~52.3M floats (~210 MB)
    float* ws = (float*)d_ws;
    long o = 0;
    float* X   = ws + o; o += (long)NTOK * 64;        // activation (B,T,N,64)
    float* T1  = ws + o; o += (long)NTOK * 64;        // hconv tmp / fuse out / gru seq0
    float* HX  = ws + o; o += (long)NTOK * 64;        // hconv out / gru input (B,N,T,64)
    float* ON  = ws + o; o += (long)NTOK * 64;        // kan_nei out / gru seq1
    float* OS  = ws + o; o += (long)NTOK * 64;        // kan_sem out
    float* h0  = ws + o; o += (long)NSEQ * 64;
    float* h1  = ws + o; o += (long)NSEQ * 64;
    float* gi  = ws + o; o += (long)NSEQ * 192;
    float* gh  = ws + o; o += (long)NSEQ * 192;
    float* s1n = ws + o; o += HE1;
    float* s1s = ws + o; o += HE2;
    float* s2n = ws + o; o += HN;
    float* s2s = ws + o; o += HN;

    dim3 blk(256);
    // hypergraph normalization scales (constant for the whole model)
    edge_scale<<<dim3((HE1 + 255) / 256), blk, 0, stream>>>(H_nei, W_nei, s1n, HN, HE1);
    edge_scale<<<dim3((HE2 + 255) / 256), blk, 0, stream>>>(H_sem, W_sem, s1s, HN, HE2);
    node_scale<<<dim3(12), blk, 0, stream>>>(H_nei, W_nei, s2n, HN, HE1);
    node_scale<<<dim3(12), blk, 0, stream>>>(H_sem, W_sem, s2s, HN, HE2);

    // input projection (K=2: pure VALU, WMMA not worth it)
    in_proj<<<dim3(36000), blk, 0, stream>>>(x, in_W, in_b, X);

    // encoder layers
    run_layer(enc0, H_nei, s1n, s2n, H_sem, s1s, s2s, X, T1, HX, ON, OS, stream);
    run_layer(enc1, H_nei, s1n, s2n, H_sem, s1s, s2s, X, T1, HX, ON, OS, stream);

    // encoder GRU (h0,h1 become h_n, reused directly as decoder init)
    transpose_btnc<<<dim3(36000), blk, 0, stream>>>(X, HX);
    run_gru2(eg0, eg1, HX, T1, ON, h0, h1, gi, gh, /*zero_h=*/true, stream);

    // decoder: broadcast last timestep, then layers
    bcast_last<<<dim3(36000), blk, 0, stream>>>(X);
    run_layer(dec0, H_nei, s1n, s2n, H_sem, s1s, s2s, X, T1, HX, ON, OS, stream);
    run_layer(dec1, H_nei, s1n, s2n, H_sem, s1s, s2s, X, T1, HX, ON, OS, stream);

    // decoder GRU seeded with encoder h_n; final sequence lands in ON
    transpose_btnc<<<dim3(36000), blk, 0, stream>>>(X, HX);
    run_gru2(dg0, dg1, HX, T1, ON, h0, h1, gi, gh, /*zero_h=*/false, stream);

    // output projection -> (B,T,N,1)
    out_proj<<<dim3(563), blk, 0, stream>>>(ON, out_W, out_b, (float*)d_out);
}
